// PatchSelfAttentionBlock_23441931502286
// MI455X (gfx1250) — matrix operand order repaired and checked by reading, not compiled, not run on hardware
//
#include <hip/hip_runtime.h>
#include <hip/hip_bf16.h>

typedef __attribute__((ext_vector_type(16))) _Float16 v16h;
typedef __attribute__((ext_vector_type(8)))  _Float16 h8;
typedef __attribute__((ext_vector_type(8)))  float    v8f;

#define WMMA_F16(A, B, C) \
  __builtin_amdgcn_wmma_f32_16x16x32_f16(false, (A), false, (B), (short)0, (C), false, false)

// ---------------------------------------------------------------------------
// Sizes (hardcoded to the reference problem)
// B=8, C=512, H=W=32, N=1024, heads=8, d=64
// ---------------------------------------------------------------------------

// Convert the four 512x512 f32 weight matrices to f16, concatenated
// (wq | wk | wv | wo) so the QKV GEMM can index by blockIdx.z.
__global__ __launch_bounds__(256) void prep_weights(
    const float* __restrict__ wq, const float* __restrict__ wk,
    const float* __restrict__ wv, const float* __restrict__ wo,
    _Float16* __restrict__ dst) {
  int idx = blockIdx.x * 256 + threadIdx.x;       // < 4*262144
  int which = idx >> 18;
  int off   = idx & 262143;
  const float* src = (which == 0) ? wq : (which == 1) ? wk : (which == 2) ? wv : wo;
  dst[idx] = (_Float16)src[off];
}

// x + sinusoidal 2D positional encoding, transposed to [B][N][C], f16.
__global__ __launch_bounds__(256) void prep_x(const float* __restrict__ x,
                                              _Float16* __restrict__ Xh) {
  int idx = blockIdx.x * 256 + threadIdx.x;       // < 8*1024*512
  int c = idx & 511;
  int n = (idx >> 9) & 1023;
  int b = idx >> 19;
  int hh = n >> 5, ww = n & 31;
  int cc = c & 255;                               // c<256 -> c ; c>=256 -> c-256
  float pos  = (c < 256) ? (float)hh : (float)ww;
  float expo = -(float)(cc & ~1) * (9.210340371976184f / 256.0f); // ln(1e4)/half
  float arg  = pos * __expf(expo);
  float pe   = (cc & 1) ? __cosf(arg) : __sinf(arg);
  float v = x[(b * 512 + c) * 1024 + n] + pe;
  Xh[(b * 1024 + n) * 512 + c] = (_Float16)v;
}

// ---------------------------------------------------------------------------
// Fragment loaders (this code's contiguous k order, same for A and B):
//  A: lane L -> row = L&15, grp = L>>4; element e = K[grp*16 + e]
//     (two contiguous 16B loads)
//  B: lane L -> col = L&15, grp = L>>4; element e = K[grp*16 + e]
//     when source is indexed [col][k] row-major (one contiguous 32B load)
// ---------------------------------------------------------------------------
__device__ inline v16h load_a_h(const _Float16* p) {   // p -> row base + grp*16
  h8 lo = *(const h8*)(p);
  h8 hi = *(const h8*)(p + 8);
  v16h a;
#pragma unroll
  for (int e = 0; e < 8; ++e) { a[e] = lo[e]; a[e + 8] = hi[e]; }
  return a;
}

__device__ inline v16h load_a_f32(const float* p) {    // LDS f32 -> f16 A-frag
  v8f lo = *(const v8f*)(p);
  v8f hi = *(const v8f*)(p + 8);
  v16h a;
#pragma unroll
  for (int e = 0; e < 8; ++e) { a[e] = (_Float16)lo[e]; a[e + 8] = (_Float16)hi[e]; }
  return a;
}

// ---------------------------------------------------------------------------
// QKV projection: Out[n][o] = sum_c Xh[n][c] * W[o][c]
// grid (512, 2, 3), 128 threads = 4 waves.
// Each wave computes a 16x64 tile: one A-fragment feeds 4 WMMAs per k-step.
// z==2 (V) is stored transposed: Vt[b][o][n].
// ---------------------------------------------------------------------------
__global__ __launch_bounds__(128) void qkv_gemm(
    const _Float16* __restrict__ Xh, const _Float16* __restrict__ Wqkv,
    _Float16* __restrict__ Qh, _Float16* __restrict__ Kh,
    _Float16* __restrict__ Vt) {
  const int lane = threadIdx.x & 31;
  const int wid  = threadIdx.x >> 5;
  const int row = lane & 15, grp = lane >> 4, nn = lane & 15;
  const int t  = blockIdx.x;             // 0..511
  const int b  = t >> 6;
  const int n0 = (t & 63) << 4;
  const int o0 = (blockIdx.y * 4 + wid) << 6;   // 8 wave-tiles of 64 cols
  const int z  = blockIdx.z;
  const _Float16* W = Wqkv + z * 262144;

  const _Float16* Arow = Xh + (((b << 10) + n0 + row) << 9) + grp * 16;
  const _Float16* Brow = W + ((o0 + nn) << 9) + grp * 16;   // +j*8192 per 16 cols

  v8f acc0 = {}, acc1 = {}, acc2 = {}, acc3 = {};
#pragma unroll
  for (int kk = 0; kk < 16; ++kk) {
    const int k0 = kk * 32;
    __builtin_prefetch((const void*)(Arow + k0 + 64), 0, 1);
    v16h a  = load_a_h(Arow + k0);
    v16h b0 = *(const v16h*)(Brow + k0);
    v16h b1 = *(const v16h*)(Brow + 8192 + k0);
    v16h b2 = *(const v16h*)(Brow + 16384 + k0);
    v16h b3 = *(const v16h*)(Brow + 24576 + k0);
    acc0 = WMMA_F16(a, b0, acc0);
    acc1 = WMMA_F16(a, b1, acc1);
    acc2 = WMMA_F16(a, b2, acc2);
    acc3 = WMMA_F16(a, b3, acc3);
  }

  v8f accs[4] = {acc0, acc1, acc2, acc3};
  if (z < 2) {
    _Float16* Out = (z == 0) ? Qh : Kh;
#pragma unroll
    for (int j = 0; j < 4; ++j)
#pragma unroll
      for (int r = 0; r < 8; ++r) {
        int M = r + (grp << 3);
        Out[(((b << 10) + n0 + M) << 9) + o0 + (j << 4) + nn] = (_Float16)accs[j][r];
      }
  } else {
#pragma unroll
    for (int j = 0; j < 4; ++j)
#pragma unroll
      for (int r = 0; r < 8; ++r) {
        int M = r + (grp << 3);
        Vt[(b << 19) + ((o0 + (j << 4) + nn) << 10) + n0 + M] = (_Float16)accs[j][r];
      }
  }
}

// ---------------------------------------------------------------------------
// Attention: one wave per (b, head, 16-row block). 64KB f32 LDS score block.
// ---------------------------------------------------------------------------
__global__ __launch_bounds__(32) void attn_kernel(
    const _Float16* __restrict__ Qh, const _Float16* __restrict__ Kh,
    const _Float16* __restrict__ Vt, _Float16* __restrict__ O) {
  __shared__ float S[16 * 1024];                 // 64 KB
  const int lane = threadIdx.x & 31;
  const int row = lane & 15, grp = lane >> 4, nn = lane & 15;
  const int t  = blockIdx.x & 63;
  const int hh = (blockIdx.x >> 6) & 7;
  const int b  = blockIdx.x >> 9;
  const int n0 = t << 4;

  // Q fragments (kept in registers for all 64 m-tiles)
  const _Float16* qrow = Qh + (((b << 10) + n0 + row) << 9) + (hh << 6) + grp * 16;
  v16h aq0 = load_a_h(qrow);
  v16h aq1 = load_a_h(qrow + 32);

  // S = (Q K^T) * scale  -> LDS
  for (int mt = 0; mt < 64; ++mt) {
    const _Float16* krow =
        Kh + (((b << 10) + (mt << 4) + nn) << 9) + (hh << 6) + grp * 16;
    v16h b0 = *(const v16h*)(krow);
    v16h b1 = *(const v16h*)(krow + 32);
    v8f s = {};
    s = WMMA_F16(aq0, b0, s);
    s = WMMA_F16(aq1, b1, s);
#pragma unroll
    for (int r = 0; r < 8; ++r)
      S[((r + (grp << 3)) << 10) + (mt << 4) + nn] = s[r] * 0.125f;
  }
  __syncthreads();

  // Softmax over each row of 1024; lane handles (row, half grp)
  float* Srow = S + (row << 10) + (grp << 9);
  float m = -3.0e38f;
  for (int c = 0; c < 512; ++c) m = fmaxf(m, Srow[c]);
  m = fmaxf(m, __shfl_xor(m, 16, 32));
  float sum = 0.0f;
  for (int c = 0; c < 512; ++c) {
    float e = __expf(Srow[c] - m);
    Srow[c] = e;
    sum += e;
  }
  sum += __shfl_xor(sum, 16, 32);
  float inv = 1.0f / sum;
  for (int c = 0; c < 512; ++c) Srow[c] *= inv;
  __syncthreads();

  // O = P * V : A-fragment from LDS (f32->f16) reused across the 4 d-tiles.
  const float* prow = S + (row << 10) + grp * 16;
  const _Float16* vbase =
      Vt + (b << 19) + (((hh << 6) + nn) << 10) + grp * 16;  // +dt*16384 per d-tile
  v8f acc0 = {}, acc1 = {}, acc2 = {}, acc3 = {};
  for (int kk = 0; kk < 32; ++kk) {
    const int k0 = kk * 32;
    v16h a = load_a_f32(prow + k0);
    acc0 = WMMA_F16(a, *(const v16h*)(vbase + k0),         acc0);
    acc1 = WMMA_F16(a, *(const v16h*)(vbase + 16384 + k0), acc1);
    acc2 = WMMA_F16(a, *(const v16h*)(vbase + 32768 + k0), acc2);
    acc3 = WMMA_F16(a, *(const v16h*)(vbase + 49152 + k0), acc3);
  }
  v8f accs[4] = {acc0, acc1, acc2, acc3};
#pragma unroll
  for (int dt = 0; dt < 4; ++dt)
#pragma unroll
    for (int r = 0; r < 8; ++r) {
      int M = r + (grp << 3);
      O[(((b << 10) + n0 + M) << 9) + (hh << 6) + (dt << 4) + nn] =
          (_Float16)accs[dt][r];
    }
}

// ---------------------------------------------------------------------------
// Output projection: out[b][c][n] = sum_o O[b][n][o] * wo[c][o] + bo[c]
// 16x64 tile per wave, same blocking as qkv_gemm.
// ---------------------------------------------------------------------------
__global__ __launch_bounds__(128) void proj_gemm(
    const _Float16* __restrict__ O, const _Float16* __restrict__ WOh,
    const float* __restrict__ bo, float* __restrict__ out) {
  const int lane = threadIdx.x & 31;
  const int wid  = threadIdx.x >> 5;
  const int row = lane & 15, grp = lane >> 4, nn = lane & 15;
  const int t  = blockIdx.x;
  const int b  = t >> 6;
  const int n0 = (t & 63) << 4;
  const int c0 = (blockIdx.y * 4 + wid) << 6;

  const _Float16* Arow = O + (((b << 10) + n0 + row) << 9) + grp * 16;
  const _Float16* Brow = WOh + ((c0 + nn) << 9) + grp * 16;

  v8f acc0 = {}, acc1 = {}, acc2 = {}, acc3 = {};
#pragma unroll
  for (int kk = 0; kk < 16; ++kk) {
    const int k0 = kk * 32;
    v16h a  = load_a_h(Arow + k0);
    acc0 = WMMA_F16(a, *(const v16h*)(Brow + k0),         acc0);
    acc1 = WMMA_F16(a, *(const v16h*)(Brow + 8192 + k0),  acc1);
    acc2 = WMMA_F16(a, *(const v16h*)(Brow + 16384 + k0), acc2);
    acc3 = WMMA_F16(a, *(const v16h*)(Brow + 24576 + k0), acc3);
  }
  v8f accs[4] = {acc0, acc1, acc2, acc3};
#pragma unroll
  for (int j = 0; j < 4; ++j) {
    float bias = bo[c0 + (j << 4) + nn];
#pragma unroll
    for (int r = 0; r < 8; ++r) {
      int M = r + (grp << 3);
      out[(b << 19) + ((c0 + (j << 4) + nn) << 10) + n0 + M] = accs[j][r] + bias;
    }
  }
}

// ---------------------------------------------------------------------------
extern "C" void kernel_launch(void* const* d_in, const int* in_sizes, int n_in,
                              void* d_out, int out_size, void* d_ws, size_t ws_size,
                              hipStream_t stream) {
  const float* x  = (const float*)d_in[0];
  const float* wq = (const float*)d_in[1];
  const float* wk = (const float*)d_in[2];
  const float* wv = (const float*)d_in[3];
  const float* wo = (const float*)d_in[4];
  const float* bo = (const float*)d_in[5];
  float* out = (float*)d_out;

  _Float16* wsH  = (_Float16*)d_ws;
  _Float16* WQKV = wsH;                       // 3 * 262144
  _Float16* WOh  = wsH + 786432;              // 262144
  _Float16* Xh   = wsH + 1048576;             // 8*1024*512
  _Float16* Qh   = wsH + 5242880;
  _Float16* Kh   = wsH + 9437184;
  _Float16* Vt   = wsH + 13631488;            // transposed [B][512][1024]
  _Float16* Oh   = Xh;                        // Xh dead after QKV -> reuse for O

  prep_weights<<<4096, 256, 0, stream>>>(wq, wk, wv, wo, wsH);
  prep_x<<<16384, 256, 0, stream>>>(x, Xh);
  qkv_gemm<<<dim3(512, 2, 3), 128, 0, stream>>>(Xh, WQKV, Qh, Kh, Vt);
  attn_kernel<<<4096, 32, 0, stream>>>(Qh, Kh, Vt, Oh);
  proj_gemm<<<dim3(512, 2), 128, 0, stream>>>(Oh, WOh, bo, out);
}
